// AttnBlock_17824114279063
// MI455X (gfx1250) — compile-verified, matrix operand order checked by reading, not run on hardware
//
#include <hip/hip_runtime.h>

typedef __attribute__((ext_vector_type(16))) _Float16 v16h;
typedef __attribute__((ext_vector_type(8)))  _Float16 v8h;
typedef __attribute__((ext_vector_type(8)))  float    v8f;

#define B_DIM  2
#define C_DIM  256
#define N_DIM  8192
#define NGROUP 32                // GroupNorm groups
#define NTILE  (N_DIM / 16)      // 512 row tiles per batch
#define KSTEPS (C_DIM / 32)      // 8 WMMA K-steps over channels
#define CTILES (C_DIM / 16)      // 16 column tiles over channels
#define JPAIRS (N_DIM / 32)      // 256 key-pair tiles of 32 keys

// ---------------------------------------------------------------------------
// WMMA helpers (CDNA5 v_wmma_f32_16x16x32_f16, wave32)
// ---------------------------------------------------------------------------
__device__ __forceinline__ v8f wmma16(v16h a, v16h b, v8f c) {
  return __builtin_amdgcn_wmma_f32_16x16x32_f16(false, a, false, b, (short)0, c,
                                                false, false);
}

// Load an A/B fragment from a row-major fp16 matrix (global or LDS).
// Per ISA layout: lane half selects K base (0/8); elements 0..7 -> K=kb..kb+7,
// elements 8..15 -> K=kb+16..kb+23.  Two 16-byte loads per lane.
__device__ __forceinline__ v16h load_frag16(const _Float16* base, int ld,
                                            int row0, int k0, int lane) {
  int r  = row0 + (lane & 15);
  int kb = (lane >> 4) << 3;                 // 0 or 8
  const _Float16* p = base + (size_t)r * ld + (k0 + kb);
  v8h lo = *(const v8h*)(p);
  v8h hi = *(const v8h*)(p + 16);
  v16h out;
#pragma unroll
  for (int i = 0; i < 8; ++i) { out[i] = lo[i]; out[i + 8] = hi[i]; }
  return out;
}

// Async global->LDS copy of one 16-byte packet (CDNA5, tracked by ASYNCcnt).
__device__ __forceinline__ void async_copy_b128(unsigned lds_off,
                                                const void* gaddr) {
  asm volatile("global_load_async_to_lds_b128 %0, %1, off"
               :: "v"(lds_off), "v"(gaddr) : "memory");
}

// ---------------------------------------------------------------------------
// Kernel 1: GroupNorm statistics.  One block per (batch, group); the 8
// channels of a group are contiguous, so this is a linear 64K-float reduction.
// ---------------------------------------------------------------------------
__global__ void gn_stats_kernel(const float* __restrict__ x,
                                float* __restrict__ stats) {
  __shared__ float rs[256], rq[256];
  int bg = blockIdx.x;                                  // 0..63
  const float* p = x + (size_t)bg * (8u * N_DIM);
  float s = 0.f, q = 0.f;
  for (int i = threadIdx.x; i < 8 * N_DIM; i += 256) {
    float v = p[i]; s += v; q += v * v;
  }
  rs[threadIdx.x] = s; rq[threadIdx.x] = q;
  __syncthreads();
  for (int st = 128; st > 0; st >>= 1) {
    if ((int)threadIdx.x < st) {
      rs[threadIdx.x] += rs[threadIdx.x + st];
      rq[threadIdx.x] += rq[threadIdx.x + st];
    }
    __syncthreads();
  }
  if (threadIdx.x == 0) {
    const float inv = 1.0f / (8.0f * N_DIM);
    float mean = rs[0] * inv;
    float var  = rq[0] * inv - mean * mean;
    stats[bg * 2 + 0] = mean;
    stats[bg * 2 + 1] = rsqrtf(var + 1e-5f);
  }
}

// ---------------------------------------------------------------------------
// Kernel 2: apply GroupNorm, cast fp16, transpose to position-major [B,N,C].
// ---------------------------------------------------------------------------
__global__ void gn_apply_kernel(const float* __restrict__ x,
                                const float* __restrict__ stats,
                                const float* __restrict__ gw,
                                const float* __restrict__ gb,
                                _Float16* __restrict__ h16) {
  int b = blockIdx.x / C_DIM;
  int c = blockIdx.x % C_DIM;
  int bg = b * NGROUP + (c >> 3);
  float mean = stats[bg * 2 + 0];
  float rstd = stats[bg * 2 + 1];
  float w  = gw[c] * rstd;
  float bi = gb[c] - mean * w;
  const float* xp = x + ((size_t)b * C_DIM + c) * N_DIM;
  _Float16* hp = h16 + (size_t)b * N_DIM * C_DIM + c;
  for (int n = threadIdx.x; n < N_DIM; n += blockDim.x)
    hp[(size_t)n * C_DIM] = (_Float16)(xp[n] * w + bi);
}

// ---------------------------------------------------------------------------
// Kernel 3: weights f32 -> f16 (wq,wk,wv,wp packed consecutively).
// ---------------------------------------------------------------------------
__global__ void wconv_kernel(const float* __restrict__ wq,
                             const float* __restrict__ wk,
                             const float* __restrict__ wv,
                             const float* __restrict__ wp,
                             _Float16* __restrict__ w16) {
  int idx = blockIdx.x * 256 + threadIdx.x;     // 0 .. 4*65536-1
  int m = idx >> 16, rem = idx & 65535;
  const float* src = (m == 0) ? wq : (m == 1) ? wk : (m == 2) ? wv : wp;
  w16[idx] = (_Float16)src[rem];
}

// ---------------------------------------------------------------------------
// Kernel 4: QKV projection.  One wave = 16-position tile x 256 out channels
// for one of {Q,K,V}.  Q is pre-scaled by 1/sqrt(C) (exact power of two).
// Q,K stored [N,C]; V stored transposed [C,N].
// ---------------------------------------------------------------------------
__global__ __launch_bounds__(256, 1)
void qkv_kernel(const _Float16* __restrict__ h16,
                const _Float16* __restrict__ w16,
                const float* __restrict__ bq,
                const float* __restrict__ bk,
                const float* __restrict__ bv,
                _Float16* __restrict__ Qn,
                _Float16* __restrict__ Kn,
                _Float16* __restrict__ Vt) {
  int wave = blockIdx.x * 8 + (threadIdx.x >> 5);
  int lane = threadIdx.x & 31;
  int m = wave % 3;                  // 0=q 1=k 2=v
  int t = wave / 3;
  int b = t >> 9;
  int n0 = (t & 511) << 4;

  const _Float16* A = h16 + (size_t)b * N_DIM * C_DIM;
  const _Float16* W = w16 + (size_t)m * C_DIM * C_DIM;

  v8f acc[CTILES];
#pragma unroll
  for (int ct = 0; ct < CTILES; ++ct) acc[ct] = {};

  for (int kk = 0; kk < KSTEPS; ++kk) {
    v16h a = load_frag16(A, C_DIM, n0, kk * 32, lane);
#pragma unroll
    for (int ct = 0; ct < CTILES; ++ct) {
      v16h bf = load_frag16(W, C_DIM, ct * 16, kk * 32, lane);
      acc[ct] = wmma16(a, bf, acc[ct]);
    }
  }

  int col = lane & 15, rowb = (lane >> 4) << 3;
  const float* bias = (m == 0) ? bq : (m == 1) ? bk : bv;
  const float postscale = (m == 0) ? 0.0625f : 1.0f;   // fold 1/sqrt(256) into Q
#pragma unroll
  for (int ct = 0; ct < CTILES; ++ct) {
    int o = ct * 16 + col;
    float bi = bias[o];
#pragma unroll
    for (int r = 0; r < 8; ++r) {
      int n = n0 + rowb + r;
      float v = (acc[ct][r] + bi) * postscale;
      if (m == 2)
        Vt[(size_t)b * C_DIM * N_DIM + (size_t)o * N_DIM + n] = (_Float16)v;
      else {
        _Float16* dst = (m == 0) ? Qn : Kn;
        dst[(size_t)b * N_DIM * C_DIM + (size_t)n * C_DIM + o] = (_Float16)v;
      }
    }
  }
}

// ---------------------------------------------------------------------------
// Kernel 5: flash-style attention with double-buffered async K/V staging.
// Block = 8 waves = 128 query rows (one batch).  All waves stream the same
// 32-key tiles, so K (32x256) and V (256x32) slices are copied once per block
// into LDS with global_load_async_to_lds_b128 (ASYNCcnt), prefetching tile
// jp+1 while computing tile jp.  Each wave keeps its 16x256 O accumulator and
// all 8 Q A-fragments in VGPRs; P is re-fragmented through per-wave LDS.
// ---------------------------------------------------------------------------
__global__ __launch_bounds__(256, 1)
void attn_kernel(const _Float16* __restrict__ Qn,
                 const _Float16* __restrict__ Kn,
                 const _Float16* __restrict__ Vt,
                 _Float16* __restrict__ An) {
  __shared__ _Float16 Ks[2][32][C_DIM];      // 2 x 16 KB
  __shared__ _Float16 Vs[2][C_DIM][32];      // 2 x 16 KB
  __shared__ _Float16 Plds[8][16][32];       // per-wave P staging, 8 KB

  int wid  = threadIdx.x >> 5;
  int lane = threadIdx.x & 31;
  int b  = blockIdx.x >> 6;                  // 64 blocks per batch
  int n0 = (blockIdx.x & 63) * 128 + wid * 16;
  int col = lane & 15, rowb = (lane >> 4) << 3;

  const _Float16* Qb = Qn + (size_t)b * N_DIM * C_DIM;
  const _Float16* Kb = Kn + (size_t)b * N_DIM * C_DIM;
  const _Float16* Vb = Vt + (size_t)b * C_DIM * N_DIM;
  _Float16* Pw = &Plds[wid][0][0];

  // --- cooperative async staging of one 32-key K/V slice (8 B128 per thread)
  auto stage = [&](int buf, int key0) {
    const char* srcK = (const char*)(Kb + (size_t)key0 * C_DIM);
    unsigned    dstK = (unsigned)(uintptr_t)&Ks[buf][0][0];
#pragma unroll
    for (int i = 0; i < 4; ++i) {            // 16 KB contiguous
      unsigned off = (threadIdx.x + i * 256) * 16u;
      async_copy_b128(dstK + off, srcK + off);
    }
    unsigned dstV = (unsigned)(uintptr_t)&Vs[buf][0][0];
#pragma unroll
    for (int i = 0; i < 4; ++i) {            // 256 rows x 64 B
      int idx = threadIdx.x + i * 256;
      int row = idx >> 2, seg = idx & 3;
      const char* src = (const char*)(Vb + (size_t)row * N_DIM + key0) + seg * 16;
      async_copy_b128(dstV + (unsigned)(row * 64 + seg * 16), src);
    }
  };

  v16h qa[KSTEPS];
#pragma unroll
  for (int kk = 0; kk < KSTEPS; ++kk)
    qa[kk] = load_frag16(Qb, C_DIM, n0, kk * 32, lane);

  v8f oacc[CTILES];
#pragma unroll
  for (int ct = 0; ct < CTILES; ++ct) oacc[ct] = {};
  float mrow[8], lrow[8];
#pragma unroll
  for (int r = 0; r < 8; ++r) { mrow[r] = -3.0e30f; lrow[r] = 0.f; }

  stage(0, 0);

  for (int jp = 0; jp < JPAIRS; ++jp) {
    int buf = jp & 1;
    if (jp + 1 < JPAIRS) {
      stage(buf ^ 1, (jp + 1) * 32);
      asm volatile("s_wait_asynccnt 0x8" ::: "memory");  // current buf complete
    } else {
      asm volatile("s_wait_asynccnt 0x0" ::: "memory");
    }
    __syncthreads();                          // buf visible to all waves

    // ---- S = Q*K^T for 32 keys (two 16x16 fragments; scale pre-folded) ----
    const _Float16* Kt = &Ks[buf][0][0];
    v8f s0 = {}, s1 = {};
#pragma unroll
    for (int kk = 0; kk < KSTEPS; ++kk) {
      v16h b0 = load_frag16(Kt, C_DIM, 0, kk * 32, lane);
      s0 = wmma16(qa[kk], b0, s0);
      v16h b1 = load_frag16(Kt, C_DIM, 16, kk * 32, lane);
      s1 = wmma16(qa[kk], b1, s1);
    }

    // ---- online softmax update per row (rows = rowb..rowb+7 per lane) ----
#pragma unroll
    for (int r = 0; r < 8; ++r) {
      float v0 = s0[r], v1 = s1[r];
      float mx = fmaxf(v0, v1);
      mx = fmaxf(mx, __shfl_xor(mx, 1, 32));
      mx = fmaxf(mx, __shfl_xor(mx, 2, 32));
      mx = fmaxf(mx, __shfl_xor(mx, 4, 32));
      mx = fmaxf(mx, __shfl_xor(mx, 8, 32));
      float mn = fmaxf(mrow[r], mx);
      float alpha = __expf(mrow[r] - mn);
      float p0 = __expf(v0 - mn), p1 = __expf(v1 - mn);
      float rs = p0 + p1;
      rs += __shfl_xor(rs, 1, 32);
      rs += __shfl_xor(rs, 2, 32);
      rs += __shfl_xor(rs, 4, 32);
      rs += __shfl_xor(rs, 8, 32);
      lrow[r] = lrow[r] * alpha + rs;
      mrow[r] = mn;
#pragma unroll
      for (int ct = 0; ct < CTILES; ++ct) oacc[ct][r] *= alpha;
      Plds[wid][rowb + r][col]      = (_Float16)p0;
      Plds[wid][rowb + r][16 + col] = (_Float16)p1;
    }

    // same-wave LDS ops are in-order; drain so P A-frag reads see the stores
    asm volatile("s_wait_dscnt 0" ::: "memory");

    // ---- O += P * V ----
    v16h pa = load_frag16(Pw, 32, 0, 0, lane);
    const _Float16* Vsb = &Vs[buf][0][0];
#pragma unroll
    for (int ct = 0; ct < CTILES; ++ct) {
      v16h vb = load_frag16(Vsb, 32, ct * 16, 0, lane);
      oacc[ct] = wmma16(pa, vb, oacc[ct]);
    }

    __syncthreads();                          // all waves done with buf
  }

  // ---- finalize: O /= l, write fp16 [N,C] ----
  _Float16* Ab = An + (size_t)b * N_DIM * C_DIM;
#pragma unroll
  for (int ct = 0; ct < CTILES; ++ct) {
#pragma unroll
    for (int r = 0; r < 8; ++r) {
      float v = oacc[ct][r] * (1.0f / lrow[r]);
      Ab[(size_t)(n0 + rowb + r) * C_DIM + ct * 16 + col] = (_Float16)v;
    }
  }
}

// ---------------------------------------------------------------------------
// Kernel 6: output projection + bias + residual, scattered back to [B,C,N] f32.
// ---------------------------------------------------------------------------
__global__ __launch_bounds__(256, 1)
void proj_kernel(const _Float16* __restrict__ An,
                 const _Float16* __restrict__ w16,
                 const float* __restrict__ bp,
                 const float* __restrict__ x,
                 float* __restrict__ out) {
  int wave = blockIdx.x * 8 + (threadIdx.x >> 5);
  int lane = threadIdx.x & 31;
  int b = wave >> 9;
  int n0 = (wave & 511) << 4;

  const _Float16* A = An + (size_t)b * N_DIM * C_DIM;
  const _Float16* W = w16 + (size_t)3 * C_DIM * C_DIM;   // wp16

  v8f acc[CTILES];
#pragma unroll
  for (int ct = 0; ct < CTILES; ++ct) acc[ct] = {};
  for (int kk = 0; kk < KSTEPS; ++kk) {
    v16h a = load_frag16(A, C_DIM, n0, kk * 32, lane);
#pragma unroll
    for (int ct = 0; ct < CTILES; ++ct) {
      v16h bf = load_frag16(W, C_DIM, ct * 16, kk * 32, lane);
      acc[ct] = wmma16(a, bf, acc[ct]);
    }
  }

  int col = lane & 15, rowb = (lane >> 4) << 3;
#pragma unroll
  for (int ct = 0; ct < CTILES; ++ct) {
    int o = ct * 16 + col;
    float bi = bp[o];
#pragma unroll
    for (int r = 0; r < 8; ++r) {
      int n = n0 + rowb + r;
      size_t idx = ((size_t)b * C_DIM + o) * N_DIM + n;
      out[idx] = x[idx] + bi + acc[ct][r];
    }
  }
}

// ---------------------------------------------------------------------------
extern "C" void kernel_launch(void* const* d_in, const int* in_sizes, int n_in,
                              void* d_out, int out_size, void* d_ws,
                              size_t ws_size, hipStream_t stream) {
  const float* x   = (const float*)d_in[0];
  const float* gnw = (const float*)d_in[1];
  const float* gnb = (const float*)d_in[2];
  const float* wq  = (const float*)d_in[3];
  const float* bq  = (const float*)d_in[4];
  const float* wk  = (const float*)d_in[5];
  const float* bk  = (const float*)d_in[6];
  const float* wv  = (const float*)d_in[7];
  const float* bv  = (const float*)d_in[8];
  const float* wp  = (const float*)d_in[9];
  const float* bp  = (const float*)d_in[10];
  float* out = (float*)d_out;

  char* ws = (char*)d_ws;
  float*    stats = (float*)ws;                              // 512 B
  _Float16* w16   = (_Float16*)(ws + 1024);                  // 512 KB
  _Float16* h16   = (_Float16*)(ws + 1024 + 4 * C_DIM * C_DIM * 2);
  const size_t BNC = (size_t)B_DIM * N_DIM * C_DIM;          // 4 Mi elems
  _Float16* Qn = h16 + BNC;
  _Float16* Kn = Qn + BNC;
  _Float16* Vt = Kn + BNC;
  _Float16* An = Vt + BNC;

  gn_stats_kernel<<<B_DIM * NGROUP, 256, 0, stream>>>(x, stats);
  gn_apply_kernel<<<B_DIM * C_DIM, 256, 0, stream>>>(x, stats, gnw, gnb, h16);
  wconv_kernel<<<(4 * C_DIM * C_DIM) / 256, 256, 0, stream>>>(wq, wk, wv, wp, w16);
  qkv_kernel<<<(B_DIM * NTILE * 3) / 8, 256, 0, stream>>>(h16, w16, bq, bk, bv,
                                                          Qn, Kn, Vt);
  attn_kernel<<<(B_DIM * NTILE) / 8, 256, 0, stream>>>(Qn, Kn, Vt, An);
  proj_kernel<<<(B_DIM * NTILE) / 8, 256, 0, stream>>>(An, w16, bp, x, out);
}
